// MultiheadAttention_66331474920137
// MI455X (gfx1250) — matrix operand order repaired and checked by reading, not compiled, not run on hardware
//
#include <hip/hip_runtime.h>
#include <hip/hip_bf16.h>

// ---------------- problem constants (from reference) ----------------
#define BB      2
#define LQ      512
#define LKV     4096
#define DIM     512     // Q_DIM = KV_DIM = QK_OUT = V_OUT = OUT_DIM
#define NH      8
#define NE      10
#define TOPK    5
#define HEADD   64      // DIM / NH
#define SCALE   0.125f  // 1/sqrt(64)

typedef __attribute__((ext_vector_type(16))) __bf16 v16bf;
typedef __attribute__((ext_vector_type(8)))  __bf16 v8bf;
typedef __attribute__((ext_vector_type(8)))  float  v8f;
typedef __attribute__((ext_vector_type(4)))  float  v4f;
typedef __attribute__((ext_vector_type(4)))  unsigned int u32x4;
typedef __attribute__((ext_vector_type(8)))  int    i32x8;
typedef __attribute__((ext_vector_type(4)))  int    i32x4;

#if __has_builtin(__builtin_amdgcn_tensor_load_to_lds) && __has_builtin(__builtin_amdgcn_s_wait_tensorcnt)
#define USE_TDM 1
#else
#define USE_TDM 0
#endif

// ---------------------------------------------------------------------
// fragment loaders (bf16 WMMA operands, contiguous k order used by this code, wave32)
//   A 16x32: row M = lane&15; K = j + 16*(lane>=16)
//   B 32x16: col N = lane&15; K = j + 16*(lane>=16)  (needs K-major source)
// ---------------------------------------------------------------------
__device__ __forceinline__ v16bf load_frag_a(const __bf16* p, int half) {
    v8bf lo = *(const v8bf*)(p + 16 * half);
    v8bf hi = *(const v8bf*)(p + 16 * half + 8);
    v16bf r;
#pragma unroll
    for (int j = 0; j < 8; ++j) { r[j] = lo[j]; r[j + 8] = hi[j]; }
    return r;
}

// p points at 16 contiguous bf16 (base + 16*half already applied)
__device__ __forceinline__ v16bf load_frag_b(const __bf16* p) {
    v8bf lo = *(const v8bf*)p;
    v8bf hi = *(const v8bf*)(p + 8);
    v16bf r;
#pragma unroll
    for (int j = 0; j < 8; ++j) { r[j] = lo[j]; r[j + 8] = hi[j]; }
    return r;
}

__device__ __forceinline__ v8f wmma_bf16(v16bf a, v16bf b, v8f c) {
    return __builtin_amdgcn_wmma_f32_16x16x32_bf16(false, a, false, b,
                                                   (short)0, c, false, false);
}

#if USE_TDM
// Issue one TDM copy: 32(K) x 16(cols) bf16 tile, K contiguous in memory
// (row stride = K elements), into LDS at lds_addr with 16B padding after
// every 64B row (pad_interval=16 DWORDs -> code 3, pad_amount=4 DWORDs ->
// code 3), giving an 80B LDS row stride (bank-conflict-free b128 reads).
__device__ __forceinline__ void tdm_load_tile(unsigned lds_addr, const __bf16* gsrc, int K) {
    unsigned long long ga = (unsigned long long)(uintptr_t)gsrc;
    u32x4 g0;
    g0[0] = 1u;                                    // count=1 (valid user D#)
    g0[1] = lds_addr;                              // LDS byte address
    g0[2] = (unsigned)ga;                          // global addr [31:0]
    g0[3] = (unsigned)((ga >> 32) & 0x01FFFFFFu)   // global addr [56:32]
          | (2u << 30);                            // type = 2 ("image")
    i32x8 g1;
    g1[0] = (int)((1u << 16)    // data_size = 1 (2 bytes)
                | (1u << 20)    // pad_enable
                | (3u << 22)    // pad_interval: 16 DWORDs
                | (3u << 25));  // pad_amount:    4 DWORDs
    g1[1] = (int)(((unsigned)K & 0xFFFFu) << 16);      // tensor_dim0 lo16
    g1[2] = (int)(((unsigned)K >> 16) & 0xFFFFu)       // tensor_dim0 hi16
          | (int)(0xFFFFu << 16);                      // tensor_dim1 lo16 (large)
    g1[3] = (int)(32u << 16);                          // tile_dim0 = 32
    g1[4] = 16;                                        // tile_dim1 = 16
    g1[5] = K;                                         // tensor_dim0_stride lo32
    g1[6] = 0;
    g1[7] = 0;
    i32x4 z4 = {0, 0, 0, 0};                           // groups 2/3 unused (2D)
    i32x8 z8 = {0, 0, 0, 0, 0, 0, 0, 0};               // extra group (6-arg form)
    __builtin_amdgcn_tensor_load_to_lds(g0, g1, z4, z4, z8, 0);
}
#endif

// =====================================================================
// 1) f32 -> bf16 elementwise convert
// =====================================================================
__global__ void f2b_kernel(const float* __restrict__ in, __bf16* __restrict__ out, int n) {
    int i = blockIdx.x * blockDim.x + threadIdx.x;
    if (i < n) out[i] = (__bf16)in[i];
}

// =====================================================================
// 1b) expert-weight transpose + convert: w[E][K][O] -> wt[E][O][K] (bf16)
// =====================================================================
__global__ void wtrans_kernel(const float* __restrict__ w, __bf16* __restrict__ wt,
                              int K, int O) {
    int idx = blockIdx.x * blockDim.x + threadIdx.x;   // over E*O*K, k fastest
    if (idx >= NE * K * O) return;
    int k = idx % K; int r = idx / K;
    int o = r % O;   int e = r / O;
    wt[idx] = (__bf16)w[((size_t)e * K + k) * O + o];
}

// =====================================================================
// 1c) V transpose: vp[B][LKV][DIM] -> vt[(b*NH+h)*HEADD + d][LKV]
// =====================================================================
__global__ void vtrans_kernel(const __bf16* __restrict__ vp, __bf16* __restrict__ vt) {
    int idx = blockIdx.x * blockDim.x + threadIdx.x;   // tok fastest
    if (idx >= BB * DIM * LKV) return;
    int tok = idx % LKV; int r = idx / LKV;
    int d = r % HEADD;   int bh = r / HEADD;
    int b = bh / NH, h = bh % NH;
    vt[idx] = vp[((size_t)b * LKV + tok) * DIM + h * HEADD + d];
}

// =====================================================================
// 2) gate kernel: one wave per token.
// =====================================================================
__global__ void gate_kernel(const float* __restrict__ x, const float* __restrict__ gw,
                            const float* __restrict__ gb, float* __restrict__ g,
                            int T, int D) {
    int wid  = (blockIdx.x * blockDim.x + threadIdx.x) >> 5;
    int lane = threadIdx.x & 31;
    if (wid >= T) return;
    const float* xr = x + (size_t)wid * D;

    float dots[NE];
    for (int e = 0; e < NE; ++e) {
        float p = 0.f;
        for (int i = lane; i < D; i += 32) p += xr[i] * gw[i * NE + e];
        for (int off = 16; off > 0; off >>= 1) p += __shfl_xor(p, off, 32);
        dots[e] = p + gb[e];
    }
    if (lane == 0) {
        float mx = dots[0];
        for (int e = 1; e < NE; ++e) mx = fmaxf(mx, dots[e]);
        float pr[NE], s = 0.f;
        for (int e = 0; e < NE; ++e) { pr[e] = __expf(dots[e] - mx); s += pr[e]; }
        float inv = 1.f / s;
        for (int e = 0; e < NE; ++e) pr[e] *= inv;
        bool ch[NE]; for (int e = 0; e < NE; ++e) ch[e] = false;
        float tsum = 0.f;
        for (int k = 0; k < TOPK; ++k) {
            int bi = 0; float bv = -1.f;
            for (int e = 0; e < NE; ++e) if (!ch[e] && pr[e] > bv) { bv = pr[e]; bi = e; }
            ch[bi] = true; tsum += bv;
        }
        float rinv = 1.f / tsum;
        for (int e = 0; e < NE; ++e) g[(size_t)wid * NE + e] = ch[e] ? pr[e] * rinv : 0.f;
    }
}

// =====================================================================
// 3) MoE GEMM with LDS-staged B tiles (TDM double-buffered).
//    Block = 256 threads = 8 waves; macro-tile 128 rows x 16 cols.
//    All 8 waves share the per-expert 32x16 B tiles staged in LDS.
//    LDS layout: ldsB[buf][e][col 0..15][40] bf16 (80B row stride:
//    64B payload + 16B pad -> conflict-free ds_load_b128).
//    MODE 0: bf16 output, MODE 1: f32 output.
// =====================================================================
template <int MODE>
__global__ void moe_gemm_kernel(const __bf16* __restrict__ X,   // [T][K]
                                const __bf16* __restrict__ WT,  // [E][O][K]
                                const float*  __restrict__ bias,// [E][O]
                                const float*  __restrict__ G,   // [T][E]
                                void* __restrict__ Yv,
                                int T, int K, int O) {
    __shared__ __bf16 ldsB[2][NE][16][40];

    const int ntn   = O >> 4;
    const int bm    = blockIdx.x / ntn;
    const int bn    = blockIdx.x - bm * ntn;
    const int m_blk = bm << 7;          // 128 rows per block
    const int n0    = bn << 4;
    const int tid   = threadIdx.x;
    const int w     = tid >> 5;
    const int lane  = tid & 31;
    const int half  = lane >> 4, l15 = lane & 15;
    const int col   = n0 + l15;
    const int m0w   = m_blk + (w << 4);

    const __bf16* xr = X + (size_t)(m0w + l15) * K;

#if USE_TDM
    unsigned ldsbase = (unsigned)(unsigned long long)
        (__attribute__((address_space(3))) char*)(&ldsB[0][0][0][0]);
#endif

    // stage one K-chunk (all experts) into buffer `buf`
    auto stage = [&](int buf, int k0) {
#if USE_TDM
        if (tid < 32) {
#pragma unroll
            for (int e = 0; e < NE; ++e) {
                unsigned laddr = ldsbase + (unsigned)(buf * NE * 16 * 40 * 2 + e * 16 * 40 * 2);
                const __bf16* gsrc = WT + ((size_t)e * O + n0) * K + k0;
                tdm_load_tile(laddr, gsrc, K);
            }
        }
#else
        // fallback: 640 x 16B copies spread over 256 threads
        for (int idx = tid; idx < NE * 16 * 4; idx += 256) {
            int e = idx >> 6; int rem = idx & 63;
            int r = rem >> 2; int c = rem & 3;
            const __bf16* gp = WT + ((size_t)(e * O + n0 + r)) * K + k0 + c * 8;
            *(v8bf*)&ldsB[buf][e][r][c * 8] = *(const v8bf*)gp;
        }
#endif
    };
    auto stage_sync = [&]() {
#if USE_TDM
        if (tid < 32) __builtin_amdgcn_s_wait_tensorcnt(0);
#endif
        __syncthreads();
    };

    v8f pe[NE];
#pragma unroll
    for (int e = 0; e < NE; ++e) pe[e] = {};

    const int nk = K >> 5;
    stage(0, 0);
    stage_sync();

    for (int ki = 0; ki < nk; ++ki) {
        int cur = ki & 1;
        if (ki + 1 < nk) stage(cur ^ 1, (ki + 1) << 5);

        v16bf a = load_frag_a(xr + (ki << 5), half);
#pragma unroll
        for (int e = 0; e < NE; ++e) {
            v16bf b = load_frag_b(&ldsB[cur][e][l15][16 * half]);
            pe[e] = wmma_bf16(a, b, pe[e]);
        }
        stage_sync();
    }

    v8f acc = {};
#pragma unroll
    for (int e = 0; e < NE; ++e) {
        float bcol = bias[e * O + col];
#pragma unroll
        for (int i = 0; i < 8; ++i) {
            int row = m0w + i + 8 * half;
            acc[i] += G[(size_t)row * NE + e] * (pe[e][i] + bcol);
        }
    }
#pragma unroll
    for (int i = 0; i < 8; ++i) {
        int row = m0w + i + 8 * half;
        size_t off = (size_t)row * O + col;
        if (MODE == 0) ((__bf16*)Yv)[off] = (__bf16)acc[i];
        else           ((float*)Yv)[off]  = acc[i];
    }
    (void)T;
}

// =====================================================================
// 4) attention scores: wave = 16 q-rows x 64 kv-cols (4 tiles, shared A).
// =====================================================================
__global__ void attn_score_kernel(const __bf16* __restrict__ Q,   // [B][LQ][DIM]
                                  const __bf16* __restrict__ Kp,  // [B][LKV][DIM]
                                  float* __restrict__ S) {        // [B][H][LQ][LKV]
    int wid  = (blockIdx.x * blockDim.x + threadIdx.x) >> 5;
    int lane = threadIdx.x & 31;
    const int QT = LQ / 16, KT4 = LKV / 64;
    if (wid >= BB * NH * QT * KT4) return;
    int kn = wid % KT4; int r = wid / KT4;
    int qm = r % QT; r /= QT;
    int h  = r % NH; int b = r / NH;
    int half = lane >> 4, l15 = lane & 15;
    int m0 = qm << 4, n0 = kn << 6;

    const __bf16* qrow = Q + ((size_t)b * LQ + m0 + l15) * DIM + h * HEADD;

    v8f c[4];
#pragma unroll
    for (int t = 0; t < 4; ++t) c[t] = {};

#pragma unroll
    for (int kc = 0; kc < HEADD; kc += 32) {
        v16bf a = load_frag_a(qrow + kc, half);
#pragma unroll
        for (int t = 0; t < 4; ++t) {
            const __bf16* kr = Kp + ((size_t)b * LKV + n0 + t * 16 + l15) * DIM
                                  + h * HEADD + kc + 16 * half;
            v16bf bf = load_frag_b(kr);
            c[t] = wmma_bf16(a, bf, c[t]);
        }
    }

    float* sb = S + ((size_t)(b * NH + h) * LQ) * LKV;
#pragma unroll
    for (int t = 0; t < 4; ++t)
#pragma unroll
        for (int i = 0; i < 8; ++i) {
            int row = m0 + i + 8 * half;
            sb[(size_t)row * LKV + n0 + t * 16 + l15] = c[t][i] * SCALE;
        }
}

// =====================================================================
// 5) in-place row softmax over LKV
// =====================================================================
__global__ void softmax_kernel(float* __restrict__ S, int n) {
    __shared__ float red[8];
    float* p = S + (size_t)blockIdx.x * n;
    int t = threadIdx.x, lane = t & 31, w = t >> 5;
    int nw = blockDim.x >> 5;

    float mx = -3.402823466e38f;
    for (int i = t; i < n; i += blockDim.x) mx = fmaxf(mx, p[i]);
    for (int off = 16; off; off >>= 1) mx = fmaxf(mx, __shfl_xor(mx, off, 32));
    if (lane == 0) red[w] = mx;
    __syncthreads();
    if (w == 0) {
        float m = (lane < nw) ? red[lane] : -3.402823466e38f;
        for (int off = 16; off; off >>= 1) m = fmaxf(m, __shfl_xor(m, off, 32));
        if (lane == 0) red[0] = m;
    }
    __syncthreads();
    mx = red[0];
    __syncthreads();

    float s = 0.f;
    for (int i = t; i < n; i += blockDim.x) { float e = __expf(p[i] - mx); p[i] = e; s += e; }
    for (int off = 16; off; off >>= 1) s += __shfl_xor(s, off, 32);
    if (lane == 0) red[w] = s;
    __syncthreads();
    if (w == 0) {
        float m = (lane < nw) ? red[lane] : 0.f;
        for (int off = 16; off; off >>= 1) m += __shfl_xor(m, off, 32);
        if (lane == 0) red[0] = m;
    }
    __syncthreads();
    float inv = 1.f / red[0];
    for (int i = t; i < n; i += blockDim.x) p[i] *= inv;
}

// =====================================================================
// 6) context = attn @ V: wave = 16 q-rows x full 64-wide head (4 tiles).
// =====================================================================
__global__ void attn_v_kernel(const float* __restrict__ P,    // [B][H][LQ][LKV]
                              const __bf16* __restrict__ VT,  // [B*H*HEADD][LKV]
                              float* __restrict__ Cf,         // [B][LQ][DIM]
                              __bf16* __restrict__ Cb) {
    int wid  = (blockIdx.x * blockDim.x + threadIdx.x) >> 5;
    int lane = threadIdx.x & 31;
    const int QT = LQ / 16;
    if (wid >= BB * NH * QT) return;
    int qm = wid % QT; int r = wid / QT;
    int h  = r % NH;   int b = r / NH;
    int half = lane >> 4, l15 = lane & 15;
    int m0 = qm << 4;

    const float*  prow  = P + ((size_t)(b * NH + h) * LQ + m0 + l15) * LKV;
    const __bf16* vbase = VT + (size_t)(b * NH + h) * HEADD * LKV;

    v8f c[4];
#pragma unroll
    for (int t = 0; t < 4; ++t) c[t] = {};

    for (int k0 = 0; k0 < LKV; k0 += 32) {
        const float* pa = prow + k0;
        v4f f0 = *(const v4f*)(pa + 16 * half);
        v4f f1 = *(const v4f*)(pa + 16 * half + 4);
        v4f f2 = *(const v4f*)(pa + 16 * half + 8);
        v4f f3 = *(const v4f*)(pa + 16 * half + 12);
        v16bf a;
#pragma unroll
        for (int j = 0; j < 4; ++j) {
            a[j]      = (__bf16)f0[j];
            a[j + 4]  = (__bf16)f1[j];
            a[j + 8]  = (__bf16)f2[j];
            a[j + 12] = (__bf16)f3[j];
        }
#pragma unroll
        for (int t = 0; t < 4; ++t) {
            const __bf16* vr = vbase + (size_t)(t * 16 + l15) * LKV + k0 + 16 * half;
            v16bf bf = load_frag_b(vr);
            c[t] = wmma_bf16(a, bf, c[t]);
        }
    }
#pragma unroll
    for (int t = 0; t < 4; ++t)
#pragma unroll
        for (int i = 0; i < 8; ++i) {
            int row = m0 + i + 8 * half;
            size_t off = ((size_t)b * LQ + row) * DIM + h * HEADD + t * 16 + l15;
            Cf[off] = c[t][i];
            Cb[off] = (__bf16)c[t][i];
        }
}

// =====================================================================
// host launcher
// =====================================================================
extern "C" void kernel_launch(void* const* d_in, const int* in_sizes, int n_in,
                              void* d_out, int out_size, void* d_ws, size_t ws_size,
                              hipStream_t stream) {
    const float* q    = (const float*)d_in[0];
    const float* kv   = (const float*)d_in[1];
    const float* q_gw = (const float*)d_in[2];
    const float* q_gb = (const float*)d_in[3];
    const float* q_w  = (const float*)d_in[4];
    const float* q_b  = (const float*)d_in[5];
    const float* k_gw = (const float*)d_in[6];
    const float* k_gb = (const float*)d_in[7];
    const float* k_w  = (const float*)d_in[8];
    const float* k_b  = (const float*)d_in[9];
    const float* v_gw = (const float*)d_in[10];
    const float* v_gb = (const float*)d_in[11];
    const float* v_w  = (const float*)d_in[12];
    const float* v_b  = (const float*)d_in[13];
    const float* o_gw = (const float*)d_in[14];
    const float* o_gb = (const float*)d_in[15];
    const float* o_w  = (const float*)d_in[16];
    const float* o_b  = (const float*)d_in[17];

    float* out_x    = (float*)d_out;                         // [B][LQ][DIM]
    float* out_attn = (float*)d_out + (size_t)BB * LQ * DIM; // [B][H][LQ][LKV]

    const int NQ  = BB * LQ;        // 1024 query tokens
    const int NKV = BB * LKV;       // 8192 kv tokens
    const int WSZ = NE * DIM * DIM; // expert weight elems

    // ---- bump allocator over d_ws ----
    char* ws = (char*)d_ws;
    size_t off = 0;
    auto alloc = [&](size_t bytes) -> void* {
        void* p = ws + off;
        off = (off + bytes + 255) & ~(size_t)255;
        return p;
    };
    __bf16* qb   = (__bf16*)alloc((size_t)NQ  * DIM * 2);
    __bf16* kvb  = (__bf16*)alloc((size_t)NKV * DIM * 2);
    __bf16* qwt  = (__bf16*)alloc((size_t)WSZ * 2);   // transposed [E][O][K]
    __bf16* kwt  = (__bf16*)alloc((size_t)WSZ * 2);
    __bf16* vwt  = (__bf16*)alloc((size_t)WSZ * 2);
    __bf16* owt  = (__bf16*)alloc((size_t)WSZ * 2);
    float*  gq   = (float*) alloc((size_t)NQ  * NE * 4);
    float*  gk   = (float*) alloc((size_t)NKV * NE * 4);
    float*  gv   = (float*) alloc((size_t)NKV * NE * 4);
    float*  go   = (float*) alloc((size_t)NQ  * NE * 4);
    __bf16* qp   = (__bf16*)alloc((size_t)NQ  * DIM * 2);
    __bf16* kp   = (__bf16*)alloc((size_t)NKV * DIM * 2);
    __bf16* vp   = (__bf16*)alloc((size_t)NKV * DIM * 2);
    __bf16* vt   = (__bf16*)alloc((size_t)NKV * DIM * 2);  // transposed V
    float*  ctxf = (float*) alloc((size_t)NQ  * DIM * 4);
    __bf16* ctxb = (__bf16*)alloc((size_t)NQ  * DIM * 2);
    (void)ws_size; (void)in_sizes; (void)n_in; (void)out_size;

    // ---- 1) conversions / weight transposes ----
    {
        int n;
        n = NQ * DIM;  f2b_kernel<<<(n + 255) / 256, 256, 0, stream>>>(q,  qb,  n);
        n = NKV * DIM; f2b_kernel<<<(n + 255) / 256, 256, 0, stream>>>(kv, kvb, n);
        n = WSZ;
        wtrans_kernel<<<(n + 255) / 256, 256, 0, stream>>>(q_w, qwt, DIM, DIM);
        wtrans_kernel<<<(n + 255) / 256, 256, 0, stream>>>(k_w, kwt, DIM, DIM);
        wtrans_kernel<<<(n + 255) / 256, 256, 0, stream>>>(v_w, vwt, DIM, DIM);
        wtrans_kernel<<<(n + 255) / 256, 256, 0, stream>>>(o_w, owt, DIM, DIM);
    }

    // ---- 2) gates ----
    gate_kernel<<<(NQ  + 3) / 4, 128, 0, stream>>>(q,  q_gw, q_gb, gq, NQ,  DIM);
    gate_kernel<<<(NKV + 3) / 4, 128, 0, stream>>>(kv, k_gw, k_gb, gk, NKV, DIM);
    gate_kernel<<<(NKV + 3) / 4, 128, 0, stream>>>(kv, v_gw, v_gb, gv, NKV, DIM);

    // ---- 3) MoE projections (block = 128 rows x 16 cols, LDS/TDM staged) ----
    {
        int bq  = (NQ  / 128) * (DIM / 16); // 256
        int bkv = (NKV / 128) * (DIM / 16); // 2048
        moe_gemm_kernel<0><<<bq,  256, 0, stream>>>(qb,  qwt, q_b, gq, qp, NQ,  DIM, DIM);
        moe_gemm_kernel<0><<<bkv, 256, 0, stream>>>(kvb, kwt, k_b, gk, kp, NKV, DIM, DIM);
        moe_gemm_kernel<0><<<bkv, 256, 0, stream>>>(kvb, vwt, v_b, gv, vp, NKV, DIM, DIM);
    }

    // ---- 3b) transpose V for K-major B fragments ----
    {
        int n = BB * DIM * LKV;
        vtrans_kernel<<<(n + 255) / 256, 256, 0, stream>>>(vp, vt);
    }

    // ---- 4) attention scores into d_out attn region ----
    {
        int waves = BB * NH * (LQ / 16) * (LKV / 64); // 32768
        attn_score_kernel<<<(waves + 7) / 8, 256, 0, stream>>>(qp, kp, out_attn);
    }

    // ---- 5) softmax rows ----
    softmax_kernel<<<BB * NH * LQ, 256, 0, stream>>>(out_attn, LKV);

    // ---- 6) context = attn @ V ----
    {
        int waves = BB * NH * (LQ / 16); // 512
        attn_v_kernel<<<(waves + 7) / 8, 256, 0, stream>>>(out_attn, vt, ctxf, ctxb);
    }

    // ---- 7) output MoE into d_out x region ----
    gate_kernel<<<(NQ + 3) / 4, 128, 0, stream>>>(ctxf, o_gw, o_gb, go, NQ, DIM);
    {
        int bq = (NQ / 128) * (DIM / 16); // 256
        moe_gemm_kernel<1><<<bq, 256, 0, stream>>>(ctxb, owt, o_b, go, out_x, NQ, DIM, DIM);
    }
}
